// LSTMModel_27376121544973
// MI455X (gfx1250) — matrix operand order repaired and checked by reading, not compiled, not run on hardware
//
#include <hip/hip_runtime.h>
#include <hip/hip_bf16.h>

// LSTM (B=64,T=2048,I=128,H=256,L=2,O=1) for MI455X / gfx1250.
// Strategy:
//   0) convert weights f32->f16, bias sums                 (parallel, tiny)
//   1) xg0 = x @ Wih0^T + b          -> ws (f16 [T][64][1024])   WMMA GEMM
//   2) layer0 recurrence, persistent 1-WG kernel (512 thr = 16 waves ->
//      4 waves/SIMD -> 256 VGPRs/wave, no spills), h in LDS, c in regs,
//      xg[t] async-prefetched to LDS (global_load_async_to_lds_b128),
//      Whh streamed from L2 each step; 4 passes/step of 4 gate-tiles with
//      deep global_load_b128 pipelining (sched_barrier pins wmmas only).
//      writes h0_seq (f16 [T][64][256])
//   3) xg1 = h0_seq @ Wih1^T + b     -> ws (reuse xg buffer)     WMMA GEMM
//   4) layer1 recurrence, same kernel; at t=T-1 writes relu(h) f32
//   5) tiny FC: out[b] = relu_h_last[b] . fc_w + fc_b
// Workspace requirement: ~323 MiB (2 MiB weights + 256 MiB xg + 64 MiB h0_seq).

#define DEVINL __device__ __forceinline__

typedef __attribute__((ext_vector_type(16))) _Float16 v16h;
typedef __attribute__((ext_vector_type(8)))  _Float16 v8h;
typedef __attribute__((ext_vector_type(8)))  float    v8f;
typedef __attribute__((ext_vector_type(4)))  unsigned int u32x4;

// explicit global-address-space views (keep weight streaming on the
// global_load path / LOADcnt even after provenance-erasing asm)
typedef __attribute__((address_space(1))) const _Float16 gcf16;
typedef __attribute__((address_space(1))) const v16h    gcv16h;

static constexpr int Bb = 64;    // batch
static constexpr int Tt = 2048;  // time steps
static constexpr int Ii = 128;   // input features (layer 0)
static constexpr int Hh = 256;   // hidden
static constexpr int Gg = 1024;  // 4*H gate width

union FragU { v16h v; v8h h[2]; };

// A-matrix 16x32 f16 fragment from row-major [16][stride] tile.
// Layout: lanes 0-15 (M=lane) hold K in {0..15}; lanes 16-31 hold
// K in {16..31}. Two contiguous 16B loads per lane.
DEVINL v16h load_frag_a(const _Float16* base, int stride, int lane) {
    const int r = lane & 15, hf = lane >> 4;
    const _Float16* p = base + r * stride + 16 * hf;
    FragU f;
    f.h[0] = *(const v8h*)(p);        // K = 16*hf + 0..7
    f.h[1] = *(const v8h*)(p + 8);    // K = 16*hf + 8..15
    return f.v;
}

// B-matrix 32x16 f16 fragment. We hold W row-major [n][k] (g = act @ W^T so
// W's row n is B's column n). Order used here (same as load_frag_a, not the
// ISA table order): lanes 0-15 hold K=0..15 for N=lane, lanes 16-31 hold K=16..31.
// One contiguous 32B load per lane.
DEVINL v16h load_frag_b(const _Float16* base, int stride, int lane) {
    const int n = lane & 15, hf = lane >> 4;
    return *(const v16h*)(base + n * stride + 16 * hf);
}

DEVINL v16h load_frag_b_g(gcf16* base, int stride, int lane) {
    const int n = lane & 15, hf = lane >> 4;
    return *(gcv16h*)(base + n * stride + 16 * hf);
}

DEVINL v8f wmma_f16(v16h a, v16h b, v8f c) {
    // emits v_wmma_f32_16x16x32_f16
    return __builtin_amdgcn_wmma_f32_16x16x32_f16(false, a, false, b,
                                                  (short)0, c, false, false);
}

DEVINL float sigmoid_f(float x) { return 1.0f / (1.0f + __expf(-x)); }
DEVINL float tanh_f(float x)    { return 2.0f / (1.0f + __expf(-2.0f * x)) - 1.0f; }

// ---------------- small prep kernels ----------------
__global__ void cvt_f32_f16(const float* __restrict__ in,
                            _Float16* __restrict__ out, int n) {
    int i = blockIdx.x * blockDim.x + threadIdx.x;
    if (i < n) out[i] = (_Float16)in[i];
}

__global__ void bias_sum(const float* __restrict__ a, const float* __restrict__ b,
                         float* __restrict__ out, int n) {
    int i = blockIdx.x * blockDim.x + threadIdx.x;
    if (i < n) out[i] = a[i] + b[i];
}

// ---------------- input-projection GEMM ----------------
// One block per (t, n-slice of 128). Block = 256 threads = 8 waves.
// Wave w computes the 64x16 strip n0 = blockIdx.y*128 + w*16 (4 M-tiles).
template <int K, bool XF32>
__global__ __launch_bounds__(256) void gemm_xg(
    const void* __restrict__ Ain,        // XF32: x f32 [B][T][I]; else h0seq f16 [T][B][K]
    const _Float16* __restrict__ W,      // f16 [1024][K]
    const float* __restrict__ bsum,      // f32 [1024]
    _Float16* __restrict__ xg)           // f16 [T][64][1024]
{
    __shared__ _Float16 As[Bb * K];
    const int t = blockIdx.x;
    const int tid = threadIdx.x;

    if constexpr (XF32) {
        const float* x = (const float*)Ain;            // [B][T][I]
        const int n4 = Bb * K / 4;
        for (int i = tid; i < n4; i += 256) {
            const int row = i / (K / 4);
            const int c4  = i % (K / 4);
            const float4 v = ((const float4*)(x + ((size_t)row * Tt + t) * K))[c4];
            _Float16* d = &As[row * K + c4 * 4];
            d[0] = (_Float16)v.x; d[1] = (_Float16)v.y;
            d[2] = (_Float16)v.z; d[3] = (_Float16)v.w;
        }
    } else {
        const u32x4* s = (const u32x4*)((const _Float16*)Ain + (size_t)t * Bb * K);
        u32x4* d = (u32x4*)As;
        const int n16 = Bb * K * 2 / 16;
        for (int i = tid; i < n16; i += 256) d[i] = s[i];
    }
    __syncthreads();

    const int wave = tid >> 5, lane = tid & 31;
    const int n0 = blockIdx.y * 128 + wave * 16;
    v8f acc[4] = {};
#pragma unroll
    for (int kc = 0; kc < K / 32; ++kc) {
        const v16h bf = load_frag_b(W + (size_t)n0 * K + kc * 32, K, lane);
        v16h af[4];
#pragma unroll
        for (int mt = 0; mt < 4; ++mt)
            af[mt] = load_frag_a(&As[mt * 16 * K + kc * 32], K, lane);
#pragma unroll
        for (int mt = 0; mt < 4; ++mt)
            acc[mt] = wmma_f16(af[mt], bf, acc[mt]);
    }

    const int ln = lane & 15, hf = lane >> 4;
    const int nn = n0 + ln;
    const float bias = bsum[nn];
    _Float16* outt = xg + (size_t)t * Bb * Gg;
#pragma unroll
    for (int mt = 0; mt < 4; ++mt)
#pragma unroll
        for (int r = 0; r < 8; ++r) {
            const int m = mt * 16 + r + 8 * hf;          // C/D layout
            outt[(size_t)m * Gg + nn] = (_Float16)(acc[mt][r] + bias);
        }
}

// ---------------- persistent recurrence kernel ----------------
// Single workgroup, 512 threads (16 waves, 4/SIMD -> 256 VGPR budget).
// Wave (mt = w&3, jq = w>>2) owns j-tiles {jq, jq+4, jq+8, jq+12}, processed
// as 4 passes; each pass computes the 4 gate tiles (i,f,g,o) for one j-tile
// so c stays in registers. New h values are staged in registers and written
// to h_buf only after all waves finished reading it (barrier), since waves
// sharing an M-tile read all columns of h_buf.
__global__ __launch_bounds__(512) void lstm_recurrent(
    const _Float16* __restrict__ xg,    // [T][64][1024] f16 (bias included)
    const _Float16* __restrict__ Whh,   // [1024][256] f16
    _Float16* __restrict__ hseq,        // [T][64][256] f16 or nullptr
    float* __restrict__ hlast,          // [64][256] f32 (relu) or nullptr
    int T)
{
    __shared__ _Float16 h_buf[Bb * Hh];       // 32 KB
    __shared__ _Float16 xg_buf[Bb * Gg];      // 128 KB  (160 KB total LDS)

    const int tid  = threadIdx.x;
    const int wave = tid >> 5, lane = tid & 31;
    const int mt = wave & 3, jq = wave >> 2;     // jq in 0..3
    const int ln = lane & 15, hf = lane >> 4;

    for (int i = tid; i < Bb * Hh; i += 512) h_buf[i] = (_Float16)0.0f;

    const unsigned lds_xg = (unsigned)(unsigned long long)&xg_buf[0];
    const unsigned my_off = (unsigned)tid * 16u;

    // prefetch xg[0] -> LDS via CDNA5 async path (ASYNCcnt): 16 x b128/thread
    {
        const _Float16* src = xg;
#pragma unroll
        for (int c = 0; c < 16; ++c) {
            const unsigned o = my_off + (unsigned)c * 8192u;
            asm volatile("global_load_async_to_lds_b128 %0, %1, %2 offset:0"
                         :: "v"(lds_xg + o), "v"(o), "s"(src) : "memory");
        }
    }

    float cst[4][8];    // c state: 4 j-tiles x 8 rows
    float hrg[4][8];    // staged h results for deferred LDS write
#pragma unroll
    for (int p = 0; p < 4; ++p)
#pragma unroll
        for (int r = 0; r < 8; ++r) cst[p][r] = 0.0f;

    __syncthreads();

    const _Float16* Wop = Whh;
    for (int t = 0; t < T; ++t) {
        // Anti-LICM: opaque weight pointer; weights stream from L2 per step.
        asm volatile("" : "+s"(Wop));
        gcf16* Wg = (gcf16*)(unsigned long long)Wop;   // -> global_load_b128

#pragma unroll
        for (int p = 0; p < 4; ++p) {
            const int jt = p * 4 + jq;
            // ---- pass p: acc(4 gate tiles) += h @ Whh^T columns of jt ----
            v8f acc[4] = {};
#pragma unroll
            for (int kc = 0; kc < Hh / 32; ++kc) {
                const v16h af =
                    load_frag_a(&h_buf[(mt * 16) * Hh + kc * 32], Hh, lane);
                v16h bf[4];
#pragma unroll
                for (int g = 0; g < 4; ++g) {
                    const int n0 = g * Hh + jt * 16;
                    bf[g] = load_frag_b_g(Wg + (size_t)n0 * Hh + kc * 32, Hh, lane);
                }
                // pin wmmas behind their loads; allow ALU/VMEM/DS to cross so
                // the next chunk's loads can hoist above these wmmas
                __builtin_amdgcn_sched_barrier(0x497);
#pragma unroll
                for (int g = 0; g < 4; ++g)
                    acc[g] = wmma_f16(af, bf[g], acc[g]);
            }

            if (p == 0) {
                // xg[t] copy (issued at end of previous step) complete:
                // own slices (wait) + everyone's slices (barrier)
                asm volatile("s_wait_asynccnt 0" ::: "memory");
                __syncthreads();   // barrier A
            }

            // ---- gates + state update; h staged in registers ----
            const int j = jt * 16 + ln;
#pragma unroll
            for (int r = 0; r < 8; ++r) {
                const int m = mt * 16 + r + 8 * hf;
                const _Float16* xrow = &xg_buf[m * Gg];
                const float gi = acc[0][r] + (float)xrow[0 * Hh + j];
                const float gf = acc[1][r] + (float)xrow[1 * Hh + j];
                const float gc = acc[2][r] + (float)xrow[2 * Hh + j];
                const float go = acc[3][r] + (float)xrow[3 * Hh + j];
                const float iv = sigmoid_f(gi);
                const float fv = sigmoid_f(gf);
                const float gv = tanh_f(gc);
                const float ov = sigmoid_f(go);
                const float c = fv * cst[p][r] + iv * gv;
                cst[p][r] = c;
                const float hv = ov * tanh_f(c);
                hrg[p][r] = hv;
                if (hseq) hseq[((size_t)t * Bb + m) * Hh + j] = (_Float16)hv;
                if (hlast && t == T - 1) hlast[m * Hh + j] = fmaxf(hv, 0.0f);
            }
        }

        __syncthreads();   // barrier B: all h_buf / xg_buf reads done

        // ---- deferred h_buf update from registers ----
#pragma unroll
        for (int p = 0; p < 4; ++p) {
            const int j = (p * 4 + jq) * 16 + ln;
#pragma unroll
            for (int r = 0; r < 8; ++r) {
                const int m = mt * 16 + r + 8 * hf;
                h_buf[m * Hh + j] = (_Float16)hrg[p][r];
            }
        }

        // prefetch xg[t+1]; overlaps next step's first matmul pass
        if (t + 1 < T) {
            const _Float16* src = xg + (size_t)(t + 1) * Bb * Gg;
#pragma unroll
            for (int c = 0; c < 16; ++c) {
                const unsigned o = my_off + (unsigned)c * 8192u;
                asm volatile("global_load_async_to_lds_b128 %0, %1, %2 offset:0"
                             :: "v"(lds_xg + o), "v"(o), "s"(src) : "memory");
            }
        }

        __syncthreads();   // barrier C: h_buf writes visible for next step
    }
}

// ---------------- tiny FC head (O=1) ----------------
__global__ void fc_out(const float* __restrict__ hl,   // relu(h_T) [64][256]
                       const float* __restrict__ w,    // [256]
                       const float* __restrict__ b,    // [1]
                       float* __restrict__ out) {      // [64]
    const int bb = threadIdx.x;
    if (bb < Bb) {
        float s = 0.0f;
        for (int j = 0; j < Hh; ++j) s += hl[bb * Hh + j] * w[j];
        out[bb] = s + b[0];
    }
}

extern "C" void kernel_launch(void* const* d_in, const int* in_sizes, int n_in,
                              void* d_out, int out_size, void* d_ws, size_t ws_size,
                              hipStream_t stream) {
    (void)in_sizes; (void)n_in; (void)out_size; (void)ws_size;
    const float* x    = (const float*)d_in[0];
    const float* wih0 = (const float*)d_in[1];
    const float* whh0 = (const float*)d_in[2];
    const float* bih0 = (const float*)d_in[3];
    const float* bhh0 = (const float*)d_in[4];
    const float* wih1 = (const float*)d_in[5];
    const float* whh1 = (const float*)d_in[6];
    const float* bih1 = (const float*)d_in[7];
    const float* bhh1 = (const float*)d_in[8];
    const float* fcw  = (const float*)d_in[9];
    const float* fcb  = (const float*)d_in[10];

    char* ws = (char*)d_ws;
    _Float16* wih0h = (_Float16*)(ws + 0);          // 1024*128*2 = 256 KiB
    _Float16* whh0h = (_Float16*)(ws + 262144);     // 512 KiB
    _Float16* wih1h = (_Float16*)(ws + 786432);     // 512 KiB
    _Float16* whh1h = (_Float16*)(ws + 1310720);    // 512 KiB
    float*    bsum0 = (float*)(ws + 1835008);       // 4 KiB
    float*    bsum1 = (float*)(ws + 1839104);       // 4 KiB
    float*    hlast = (float*)(ws + 1843200);       // 64 KiB
    _Float16* xg    = (_Float16*)(ws + (size_t)(2u << 20));              // 256 MiB
    _Float16* h0seq = (_Float16*)(ws + (size_t)(2u << 20)
                                     + (size_t)Tt * Bb * Gg * 2);        // 64 MiB

    // weight conversion + bias sums
    cvt_f32_f16<<<(Gg * Ii + 255) / 256, 256, 0, stream>>>(wih0, wih0h, Gg * Ii);
    cvt_f32_f16<<<(Gg * Hh + 255) / 256, 256, 0, stream>>>(whh0, whh0h, Gg * Hh);
    cvt_f32_f16<<<(Gg * Hh + 255) / 256, 256, 0, stream>>>(wih1, wih1h, Gg * Hh);
    cvt_f32_f16<<<(Gg * Hh + 255) / 256, 256, 0, stream>>>(whh1, whh1h, Gg * Hh);
    bias_sum<<<(Gg + 255) / 256, 256, 0, stream>>>(bih0, bhh0, bsum0, Gg);
    bias_sum<<<(Gg + 255) / 256, 256, 0, stream>>>(bih1, bhh1, bsum1, Gg);

    // layer 0
    gemm_xg<Ii, true><<<dim3(Tt, 8), 256, 0, stream>>>(x, wih0h, bsum0, xg);
    lstm_recurrent<<<1, 512, 0, stream>>>(xg, whh0h, h0seq, nullptr, Tt);
    // layer 1
    gemm_xg<Hh, false><<<dim3(Tt, 8), 256, 0, stream>>>(h0seq, wih1h, bsum1, xg);
    lstm_recurrent<<<1, 512, 0, stream>>>(xg, whh1h, nullptr, hlast, Tt);
    // head
    fc_out<<<1, 64, 0, stream>>>(hlast, fcw, fcb, (float*)d_out);
}
